// RNNModel_31129922961928
// MI455X (gfx1250) — matrix operand order repaired and checked by reading, not compiled, not run on hardware
//
#include <hip/hip_runtime.h>
#include <hip/hip_bf16.h>

typedef __attribute__((ext_vector_type(16))) _Float16 v16h;
typedef __attribute__((ext_vector_type(8)))  float    v8f;

namespace {
constexpr int  B_SZ = 32;
constexpr int  T_SZ = 1024;
constexpr int  D_SZ = 256;
constexpr int  H_SZ = 512;
constexpr int  O_SZ = 256;
constexpr int  G4   = 4 * H_SZ;        // 2048 gate columns (G,I,F,O blocks of 512)
constexpr int  XKB0 = D_SZ / 32;       // 8  K-blocks for x part (layer 0)
constexpr int  HKB  = H_SZ / 32;       // 16 K-blocks for h part
constexpr int  KB0  = XKB0 + HKB;      // 24 total K-blocks layer 0
constexpr int  KB1  = HKB + HKB;       // 32 total K-blocks layer 1
constexpr int  NT   = G4 / 16;         // 128 gate column tiles
constexpr int  FRAG = 512;             // halves per 16x32 / 32x16 fragment (32 lanes * 16)
}

__device__ __forceinline__ v8f wmma_f16(v16h a, v16h b, v8f c) {
  // D = A(16x32 f16) * B(32x16 f16) + C(16x16 f32)
  return __builtin_amdgcn_wmma_f32_16x16x32_f16(
      /*neg_a=*/false, a, /*neg_b=*/false, b,
      /*c_mod=*/(short)0, c, /*reuse_a=*/false, /*reuse_b=*/false);
}

// ---------------------------------------------------------------------------
// Zero a u32 region (barrier counters + h ping-pong buffers) — runs each call
// so graph replays are deterministic.
__global__ void zero_kernel(unsigned* __restrict__ p, size_t nwords) {
  size_t i = (size_t)blockIdx.x * blockDim.x + threadIdx.x;
  size_t stride = (size_t)gridDim.x * blockDim.x;
  for (; i < nwords; i += stride) p[i] = 0u;
}

// ---------------------------------------------------------------------------
// Pack x [B,T,D] fp32 -> A-fragment layout fp16:
//   out[tile = t*2 + b/16][kb][lane][half], 16x32 A layout per ISA 7.12.2:
//   lanes 0-15: halves 0-7 -> K 0-7, halves 8-15 -> K 16-23
//   lanes16-31: halves 0-7 -> K 8-15, halves 8-15 -> K 24-31 ; row M = lane%16
__global__ void pack_x_kernel(const float* __restrict__ x, _Float16* __restrict__ out) {
  size_t idx = (size_t)blockIdx.x * blockDim.x + threadIdx.x;
  size_t total = (size_t)(T_SZ * 2) * XKB0 * FRAG;
  if (idx >= total) return;
  int half = (int)(idx & 15);
  int lane = (int)((idx >> 4) & 31);
  size_t rem = idx >> 9;
  int kb   = (int)(rem % XKB0);
  int tile = (int)(rem / XKB0);
  int t  = tile >> 1;
  int bt = tile & 1;
  int m  = lane & 15;
  int hi = lane >> 4;
  int kk = (half & 7) + 8 * hi + 16 * (half >> 3);
  int b  = bt * 16 + m;
  int k  = kb * 32 + kk;
  out[idx] = (_Float16)x[((size_t)b * T_SZ + t) * D_SZ + k];
}

// ---------------------------------------------------------------------------
// Pack stacked weights [Wx ; Wh] -> B-fragment layout fp16:
//   out[nTile][kb][lane][half], 32x16 B layout:
//   lanes 0-15: col N = lane, halves 0-7 -> K 0-7, 8-15 -> K 16-23 ; lanes 16-31: col N = lane-16,
//   halves 0-7 -> K 8-15, 8-15 -> K 24-31.  B[k][n] = (k < KX ? Wx[n][k] : Wh[n][k-KX]).
__global__ void pack_b_kernel(const float* __restrict__ Wx, const float* __restrict__ Wh,
                              _Float16* __restrict__ out, int KX, int totKB) {
  size_t idx = (size_t)blockIdx.x * blockDim.x + threadIdx.x;
  size_t total = (size_t)NT * totKB * FRAG;
  if (idx >= total) return;
  int half = (int)(idx & 15);
  int lane = (int)((idx >> 4) & 31);
  size_t rem = idx >> 9;
  int kb    = (int)(rem % totKB);
  int nTile = (int)(rem / totKB);
  int n  = nTile * 16 + (lane & 15);
  int kk = (half & 7) + 8 * (lane >> 4) + 16 * (half >> 3);
  int k  = kb * 32 + kk;
  float v = (k < KX) ? Wx[(size_t)n * KX + k]
                     : Wh[(size_t)n * H_SZ + (k - KX)];
  out[idx] = (_Float16)v;
}

// Combined gate bias: bx + bh, flat [2048] fp32.
__global__ void bias_kernel(const float* __restrict__ bx, const float* __restrict__ bh,
                            float* __restrict__ out) {
  int i = blockIdx.x * blockDim.x + threadIdx.x;
  if (i < G4) out[i] = bx[i] + bh[i];
}

// ---------------------------------------------------------------------------
// Persistent recurrent kernel. Grid: 32 blocks x 64 threads = 64 wave32s.
// Wave wid -> (mTile = wid/32 in {0,1}, jTile = wid%32 over H columns).
// Each wave computes the 4 gate tiles (G,I,F,O) for its 16 h-columns:
//   gates = bias + [x_t | h_{t-1}] @ [Wx ; Wh]^T    (K = 32*totKB via WMMA)
// B (weight) fragments are staged ONCE into LDS (2 waves * 4 gates * totKB
// * 1KB = 192/256 KB, fits CDNA5's 320KB-per-WG LDS), so the serial per-step
// critical path reads weights via ds_load instead of re-streaming L2.
// c lives in VGPRs across all T steps; h ping-pongs through a small global
// buffer in A-fragment layout; ONE grid barrier per step.
__global__ __launch_bounds__(64, 1)
void lstm_recur_kernel(const _Float16* __restrict__ packedX, int xKB,
                       const _Float16* __restrict__ packedB, int totKB,
                       const float* __restrict__ bias,
                       _Float16* __restrict__ hbuf,      // [2][2 tiles][HKB][32][16]
                       _Float16* __restrict__ hsOut,     // layer0: [T*2][HKB][32][16] or null
                       float*    __restrict__ hFinal,    // layer1: [32][512] fp32 or null
                       unsigned* __restrict__ bar) {
  extern __shared__ __align__(16) char smem_raw[];
  _Float16* bLds = (_Float16*)smem_raw;   // [wave][gate][totKB][32][16]

  const int lane  = threadIdx.x & 31;
  const int wib   = threadIdx.x >> 5;
  const int wid   = blockIdx.x * (blockDim.x >> 5) + wib;  // 0..63
  const int mTile = wid >> 5;                              // 0..1 (batch tile)
  const int jTile = wid & 31;                              // 0..31 (h column tile)
  const int nC    = lane & 15;                             // C-tile column
  const int hiC   = lane >> 4;                             // C-tile row-half select

  // ---- one-time global -> LDS stage of this WG's weight fragments ---------
  {
    const int fragInt4 = FRAG / 8;                 // 64 int4 per fragment
    for (int chunk = 0; chunk < 8; ++chunk) {      // 2 waves * 4 gates
      const int w  = chunk >> 2;
      const int g  = chunk & 3;
      const int jT = (blockIdx.x * (blockDim.x >> 5) + w) & 31;
      const int4* src = (const int4*)(packedB + (size_t)((g * 32 + jT) * totKB) * FRAG);
      int4* dst = (int4*)(bLds + (size_t)((w * 4 + g) * totKB) * FRAG);
      const int nElem = totKB * fragInt4;
      for (int e = threadIdx.x; e < nElem; e += blockDim.x)
        dst[e] = src[e];
    }
    __syncthreads();
  }
  const _Float16* bWave = bLds + (size_t)(wib * 4) * totKB * FRAG;

  // Per-lane gate biases for this column (constant across steps).
  float bg0 = bias[0 * H_SZ + jTile * 16 + nC];
  float bg1 = bias[1 * H_SZ + jTile * 16 + nC];
  float bg2 = bias[2 * H_SZ + jTile * 16 + nC];
  float bg3 = bias[3 * H_SZ + jTile * 16 + nC];

  // Destination scatter pattern for h -> A-fragment layout (constant per lane).
  const int j    = jTile * 16 + nC;       // h column index
  const int kb_h = j >> 5;
  const int kk   = j & 31;
  const int grp  = kk >> 3;
  const int laneP = 16 * (grp & 1);       // + m_local added per element
  const int halfP = (kk & 7) + 8 * (grp >> 1);

  v8f creg = {};                          // persistent cell state (8 rows per lane)

  const size_t hTileHalves = (size_t)HKB * FRAG;           // one batch-tile of h frags

  for (int t = 0; t < T_SZ; ++t) {
    v8f acc0, acc1, acc2, acc3;
#pragma unroll
    for (int i = 0; i < 8; ++i) { acc0[i] = bg0; acc1[i] = bg1; acc2[i] = bg2; acc3[i] = bg3; }

    // ---- x_t part of the concat GEMM -------------------------------------
    const _Float16* aX = packedX + (size_t)(t * 2 + mTile) * xKB * FRAG;
    if (t + 1 < T_SZ)  // warm L2/WGP$ for the next step's x fragments
      __builtin_prefetch(packedX + (size_t)((t + 1) * 2 + mTile) * xKB * FRAG, 0, 1);
#pragma unroll 2
    for (int kb = 0; kb < xKB; ++kb) {
      v16h a = *((const v16h*)(aX + (size_t)kb * FRAG) + lane);
      acc0 = wmma_f16(a, *((const v16h*)(bWave + (size_t)(0 * totKB + kb) * FRAG) + lane), acc0);
      acc1 = wmma_f16(a, *((const v16h*)(bWave + (size_t)(1 * totKB + kb) * FRAG) + lane), acc1);
      acc2 = wmma_f16(a, *((const v16h*)(bWave + (size_t)(2 * totKB + kb) * FRAG) + lane), acc2);
      acc3 = wmma_f16(a, *((const v16h*)(bWave + (size_t)(3 * totKB + kb) * FRAG) + lane), acc3);
    }

    // ---- h_{t-1} part of the concat GEMM ---------------------------------
    const _Float16* aH = hbuf + ((size_t)(t & 1) * 2 + mTile) * hTileHalves;
#pragma unroll 2
    for (int kb = 0; kb < HKB; ++kb) {
      v16h a = *((const v16h*)(aH + (size_t)kb * FRAG) + lane);
      const int kbb = xKB + kb;
      acc0 = wmma_f16(a, *((const v16h*)(bWave + (size_t)(0 * totKB + kbb) * FRAG) + lane), acc0);
      acc1 = wmma_f16(a, *((const v16h*)(bWave + (size_t)(1 * totKB + kbb) * FRAG) + lane), acc1);
      acc2 = wmma_f16(a, *((const v16h*)(bWave + (size_t)(2 * totKB + kbb) * FRAG) + lane), acc2);
      acc3 = wmma_f16(a, *((const v16h*)(bWave + (size_t)(3 * totKB + kbb) * FRAG) + lane), acc3);
    }

    // ---- activations, cell/hidden update, scatter h ----------------------
    _Float16* hw = hbuf + ((size_t)((t + 1) & 1) * 2 + mTile) * hTileHalves
                 + (size_t)kb_h * FRAG;
    _Float16* hs = hsOut ? hsOut + (size_t)(t * 2 + mTile) * hTileHalves
                         + (size_t)kb_h * FRAG
                         : nullptr;
#pragma unroll
    for (int i = 0; i < 8; ++i) {
      float g = tanhf(acc0[i]);
      float ig = 1.0f / (1.0f + expf(-acc1[i]));
      float fg = 1.0f / (1.0f + expf(-acc2[i]));
      float og = 1.0f / (1.0f + expf(-acc3[i]));
      float c = fg * creg[i] + ig * g;
      creg[i] = c;
      float h = og * tanhf(c);
      int m_local = i + 8 * hiC;                 // C layout: lanes16-31 hold M=i+8
      size_t pos = (size_t)(m_local + laneP) * 16 + halfP;
      _Float16 h16 = (_Float16)h;
      hw[pos] = h16;
      if (hs) hs[pos] = h16;
      if (hFinal && t == T_SZ - 1)
        hFinal[(size_t)(mTile * 16 + m_local) * H_SZ + j] = h;
    }

    // ---- one device-wide barrier per step --------------------------------
    if (t + 1 < T_SZ) {
      __threadfence();                     // release h stores to device scope
      __syncthreads();
      if (threadIdx.x == 0) {
        __hip_atomic_fetch_add(&bar[t], 1u, __ATOMIC_RELEASE, __HIP_MEMORY_SCOPE_AGENT);
        while (__hip_atomic_load(&bar[t], __ATOMIC_ACQUIRE, __HIP_MEMORY_SCOPE_AGENT)
               < gridDim.x)
          __builtin_amdgcn_s_sleep(1);
      }
      __syncthreads();
    }
  }
}

// ---------------------------------------------------------------------------
// Final FC: out[b][o] = fcb[o] + sum_j hFinal[b][j] * fcW[o][j]   (4 MFLOP)
__global__ void fc_kernel(const float* __restrict__ h, const float* __restrict__ W,
                          const float* __restrict__ bvec, float* __restrict__ out) {
  int idx = blockIdx.x * blockDim.x + threadIdx.x;
  if (idx >= B_SZ * O_SZ) return;
  int b = idx / O_SZ, o = idx % O_SZ;
  const float* hr = h + (size_t)b * H_SZ;
  const float* wr = W + (size_t)o * H_SZ;
  float s = bvec[o];
#pragma unroll 4
  for (int jj = 0; jj < H_SZ; ++jj) s += hr[jj] * wr[jj];
  out[idx] = s;
}

// ---------------------------------------------------------------------------
extern "C" void kernel_launch(void* const* d_in, const int* in_sizes, int n_in,
                              void* d_out, int out_size, void* d_ws, size_t ws_size,
                              hipStream_t stream) {
  const float* x   = (const float*)d_in[0];
  const float* Wx0 = (const float*)d_in[1];
  const float* bx0 = (const float*)d_in[2];
  const float* Wh0 = (const float*)d_in[3];
  const float* bh0 = (const float*)d_in[4];
  const float* Wx1 = (const float*)d_in[5];
  const float* bx1 = (const float*)d_in[6];
  const float* Wh1 = (const float*)d_in[7];
  const float* bh1 = (const float*)d_in[8];
  const float* fcW = (const float*)d_in[9];
  const float* fcb = (const float*)d_in[10];

  // -------- workspace carving (all offsets multiples of 256B) -------------
  char* ws = (char*)d_ws;
  size_t off = 0;
  auto take = [&](size_t bytes) {
    size_t o = off;
    off = (off + bytes + 255) & ~(size_t)255;
    return o;
  };
  const size_t off_pb0  = take((size_t)NT * KB0 * FRAG * sizeof(_Float16)); // 3 MB
  const size_t off_pb1  = take((size_t)NT * KB1 * FRAG * sizeof(_Float16)); // 4 MB
  const size_t off_bia0 = take((size_t)G4 * sizeof(float));
  const size_t off_bia1 = take((size_t)G4 * sizeof(float));
  // bar + h0 + h1 kept contiguous so one zero_kernel clears all per-call state
  const size_t off_bar  = take((size_t)2 * T_SZ * sizeof(unsigned));        // 8 KB
  const size_t off_h0   = take((size_t)2 * 2 * HKB * FRAG * sizeof(_Float16)); // 64 KB
  const size_t off_h1   = take((size_t)2 * 2 * HKB * FRAG * sizeof(_Float16)); // 64 KB
  const size_t off_hfin = take((size_t)B_SZ * H_SZ * sizeof(float));
  const size_t off_px   = take((size_t)(T_SZ * 2) * XKB0 * FRAG * sizeof(_Float16)); // 16 MB
  const size_t off_hs0  = take((size_t)(T_SZ * 2) * HKB  * FRAG * sizeof(_Float16)); // 32 MB
  if (ws_size < off) return;  // workspace too small; cannot run safely

  _Float16* pb0  = (_Float16*)(ws + off_pb0);
  _Float16* pb1  = (_Float16*)(ws + off_pb1);
  float*    bia0 = (float*)(ws + off_bia0);
  float*    bia1 = (float*)(ws + off_bia1);
  unsigned* bar  = (unsigned*)(ws + off_bar);
  _Float16* h0   = (_Float16*)(ws + off_h0);
  _Float16* h1   = (_Float16*)(ws + off_h1);
  float*    hfin = (float*)(ws + off_hfin);
  _Float16* px   = (_Float16*)(ws + off_px);
  _Float16* hs0  = (_Float16*)(ws + off_hs0);

  // 1) reset per-call state: barriers + both h ping-pong buffers (h_{-1}=0)
  {
    size_t nwords = (2 * T_SZ * sizeof(unsigned)
                   + 2 * 2 * 2 * HKB * FRAG * sizeof(_Float16)) / 4;
    zero_kernel<<<64, 256, 0, stream>>>(bar, nwords);
  }
  // 2) pack x and weights into WMMA fragment layouts
  {
    size_t total = (size_t)(T_SZ * 2) * XKB0 * FRAG;
    pack_x_kernel<<<(unsigned)((total + 255) / 256), 256, 0, stream>>>(x, px);
  }
  {
    size_t total = (size_t)NT * KB0 * FRAG;
    pack_b_kernel<<<(unsigned)((total + 255) / 256), 256, 0, stream>>>(Wx0, Wh0, pb0, D_SZ, KB0);
  }
  {
    size_t total = (size_t)NT * KB1 * FRAG;
    pack_b_kernel<<<(unsigned)((total + 255) / 256), 256, 0, stream>>>(Wx1, Wh1, pb1, H_SZ, KB1);
  }
  bias_kernel<<<(G4 + 255) / 256, 256, 0, stream>>>(bx0, bh0, bia0);
  bias_kernel<<<(G4 + 255) / 256, 256, 0, stream>>>(bx1, bh1, bia1);

  // 3) layer 0 recurrence: B staged to 192KB LDS; emits hs0 in A-frag layout
  lstm_recur_kernel<<<32, 64, (size_t)8 * KB0 * FRAG * sizeof(_Float16), stream>>>(
      px, XKB0, pb0, KB0, bia0, h0, hs0, nullptr, bar);
  // 4) layer 1 recurrence: B staged to 256KB LDS; consumes hs0 as its "x"
  lstm_recur_kernel<<<32, 64, (size_t)8 * KB1 * FRAG * sizeof(_Float16), stream>>>(
      hs0, HKB, pb1, KB1, bia1, h1, nullptr, hfin, bar + T_SZ);
  // 5) final FC
  fc_kernel<<<(B_SZ * O_SZ + 255) / 256, 256, 0, stream>>>(hfin, fcW, fcb, (float*)d_out);
}
